// MultiHeadSelfAttention_15298673509003
// MI455X (gfx1250) — compile-verified, matrix operand order checked by reading, not run on hardware
//
#include <hip/hip_runtime.h>
#include <hip/hip_bf16.h>

// ---------------------------------------------------------------------------
// MultiHeadSelfAttention for MI455X (gfx1250, wave32, WMMA bf16 -> f32 accum)
//   B=4, S=1024, D=1024, H=16, dqk=64, HD=1024
//   ~52 GFLOP vs ~100MB traffic -> compute bound -> v_wmma_f32_16x16x32_bf16
//   K-tile staging via TDM tensor_load_to_lds (+ LDS pad feature) when available
// ---------------------------------------------------------------------------

#define H_NUM 16
#define DQK   64
#define BSZ   4
#define SSEQ  1024
#define DMOD  1024
#define HDM   1024

typedef __attribute__((ext_vector_type(16))) __bf16 v16bf;
typedef __attribute__((ext_vector_type(8)))  __bf16 v8bf;
typedef __attribute__((ext_vector_type(8)))  float  v8f;
typedef __attribute__((ext_vector_type(4)))  unsigned u32x4;
typedef __attribute__((ext_vector_type(8)))  int      i32x8;
typedef __attribute__((ext_vector_type(4)))  int      i32x4;

#ifndef __has_builtin
#define __has_builtin(x) 0
#endif
#if __has_builtin(__builtin_amdgcn_tensor_load_to_lds) && \
    __has_builtin(__builtin_amdgcn_s_wait_tensorcnt)
#define HAVE_TDM 1
#else
#define HAVE_TDM 0
#endif

static __device__ __forceinline__ __bf16 f2bf(float f) {
  unsigned u = __builtin_bit_cast(unsigned, f);
  unsigned r = (u + 0x7fffu + ((u >> 16) & 1u)) >> 16;   // RNE
  unsigned short s = (unsigned short)r;
  return __builtin_bit_cast(__bf16, s);
}

static __device__ __forceinline__ v16bf cat8(v8bf lo, v8bf hi) {
  return __builtin_shufflevector(lo, hi, 0,1,2,3,4,5,6,7,8,9,10,11,12,13,14,15);
}

static __device__ __forceinline__ v8f wmma_bf16(v16bf a, v16bf b, v8f c) {
  // (neg_a, A, neg_b, B, c_mod, C, reuse_a, reuse_b)
  return __builtin_amdgcn_wmma_f32_16x16x32_bf16(false, a, false, b, (short)0, c,
                                                 false, false);
}

union BF16x16 { __bf16 e[16]; v8bf v[2]; };

// ---------------------------------------------------------------------------
// Generic tiled GEMM: C[M,N] = A[M,K] * W[K,N] + bias[N]
//   A fp32 (converted while staging) or bf16; out fp32 or bf16.
//   Block = 256 threads (8 waves). Tile 128x128, BK=32. Wave -> 64x32.
// ---------------------------------------------------------------------------
template <bool A_BF16, bool OUT_BF16>
__global__ __launch_bounds__(256)
void gemm_bias_kernel(const void* __restrict__ Av, const float* __restrict__ W,
                      const float* __restrict__ bias, void* __restrict__ Cv,
                      int M, int K, int N)
{
  __shared__ alignas(16) __bf16 sA[128 * 40];      // [m][k], stride 40 (pad)
  __shared__ alignas(16) __bf16 sB[128 * 40];      // transposed [n][k]

  const int t    = threadIdx.x;
  const int lane = t & 31, w = t >> 5;
  const int hl   = lane >> 4, lm = lane & 15;
  const int bM = blockIdx.y * 128, bN = blockIdx.x * 128;
  const int waveM = (w >> 2) * 64, waveN = (w & 3) * 32;

  v8f acc[4][2] = {};

  const int arow = t >> 1, akb = (t & 1) * 16;     // A: 128 rows x 32 k
  const int bk   = t >> 3, bnb = (t & 7) * 16;     // W: 32 k x 128 n

  for (int k0 = 0; k0 < K; k0 += 32) {
    // prefetch next k-tile while this one is consumed (global_prefetch_b8)
    if (k0 + 32 < K) {
      if constexpr (A_BF16)
        __builtin_prefetch((const __bf16*)Av + (size_t)(bM + arow) * K + k0 + 32 + akb, 0, 1);
      else
        __builtin_prefetch((const float*)Av + (size_t)(bM + arow) * K + k0 + 32 + akb, 0, 1);
      __builtin_prefetch(W + (size_t)(k0 + 32 + bk) * N + bN + bnb, 0, 1);
    }
    __syncthreads();
    // ---- stage A tile (convert fp32->bf16 if needed) ----
    if constexpr (A_BF16) {
      const __bf16* Ab = (const __bf16*)Av;
      const v8bf* src = (const v8bf*)(Ab + (size_t)(bM + arow) * K + k0 + akb);
      *(v8bf*)&sA[arow * 40 + akb]     = src[0];
      *(v8bf*)&sA[arow * 40 + akb + 8] = src[1];
    } else {
      const float* Af = (const float*)Av;
      const float4* s4 = (const float4*)(Af + (size_t)(bM + arow) * K + k0 + akb);
      float tmp[16];
      *(float4*)&tmp[0]  = s4[0]; *(float4*)&tmp[4]  = s4[1];
      *(float4*)&tmp[8]  = s4[2]; *(float4*)&tmp[12] = s4[3];
      BF16x16 u;
      #pragma unroll
      for (int j = 0; j < 16; ++j) u.e[j] = f2bf(tmp[j]);
      *(v8bf*)&sA[arow * 40 + akb]     = u.v[0];
      *(v8bf*)&sA[arow * 40 + akb + 8] = u.v[1];
    }
    // ---- stage W tile transposed: sB[n][k] ----
    {
      const float4* s4 = (const float4*)(W + (size_t)(k0 + bk) * N + bN + bnb);
      float tmp[16];
      *(float4*)&tmp[0]  = s4[0]; *(float4*)&tmp[4]  = s4[1];
      *(float4*)&tmp[8]  = s4[2]; *(float4*)&tmp[12] = s4[3];
      #pragma unroll
      for (int j = 0; j < 16; ++j) sB[(bnb + j) * 40 + bk] = f2bf(tmp[j]);
    }
    __syncthreads();

    // ---- fragments: per-lane K runs {hl*8..+7} and {16+hl*8..+7} ----
    v16bf afr[4];
    #pragma unroll
    for (int r = 0; r < 4; ++r) {
      const __bf16* p = &sA[(waveM + r * 16 + lm) * 40 + hl * 8];
      afr[r] = cat8(*(const v8bf*)p, *(const v8bf*)(p + 16));
    }
    v16bf bfr[2];
    #pragma unroll
    for (int c = 0; c < 2; ++c) {
      const __bf16* p = &sB[(waveN + c * 16 + lm) * 40 + hl * 8];
      bfr[c] = cat8(*(const v8bf*)p, *(const v8bf*)(p + 16));
    }
    #pragma unroll
    for (int r = 0; r < 4; ++r)
      #pragma unroll
      for (int c = 0; c < 2; ++c)
        acc[r][c] = wmma_bf16(afr[r], bfr[c], acc[r][c]);
  }

  // ---- epilogue: C layout = VGPR vg -> row (vg + 8*hl), col lm ----
  #pragma unroll
  for (int r = 0; r < 4; ++r)
    #pragma unroll
    for (int c = 0; c < 2; ++c) {
      const int n = bN + waveN + c * 16 + lm;
      const float bv = bias[n];
      #pragma unroll
      for (int vg = 0; vg < 8; ++vg) {
        const int m = bM + waveM + r * 16 + vg + 8 * hl;
        const float val = acc[r][c][vg] + bv;
        if constexpr (OUT_BF16)
          ((__bf16*)Cv)[(size_t)m * N + n] = f2bf(val);
        else
          ((float*)Cv)[(size_t)m * N + n] = val;
      }
    }
}

// ---------------------------------------------------------------------------
// Flash attention per (h,b): O = softmax(Q K^T * 0.5) V   (d = 64, S = 1024)
//   Block = 8 waves; each wave owns 16 query rows (128 q-rows / block).
//   Per 32-row KV tile per wave: 4 WMMA (QK^T) + 1 WMMA (row-sum) + 4 WMMA (P.V)
//   K tile staged by the Tensor Data Mover (with LDS pad) when available.
// ---------------------------------------------------------------------------
__global__ __launch_bounds__(256)
void flash_attn_kernel(const __bf16* __restrict__ Q, const __bf16* __restrict__ Kt,
                       const __bf16* __restrict__ V, __bf16* __restrict__ Ctx)
{
  __shared__ alignas(16) __bf16 sK [32 * 72];       // K tile  [kv][d], pad 8
  __shared__ alignas(16) __bf16 sVt[64 * 40];       // V tile transposed [d][kv]
  __shared__ alignas(16) __bf16 sP [8 * 16 * 40];   // per-wave P bounce [m][kv]

  const int t = threadIdx.x, lane = t & 31, w = t >> 5;
  const int hl = lane >> 4, lm = lane & 15;
  const int b = blockIdx.y >> 4, h = blockIdx.y & 15;
  const int qbase = blockIdx.x * 128 + w * 16;

  const size_t hoff = (size_t)b * SSEQ * HDM + (size_t)h * DQK;
  const __bf16* Qp = Q  + hoff;
  const __bf16* Kp = Kt + hoff;
  const __bf16* Vp = V  + hoff;
  __bf16*       Cp = Ctx + hoff;

  // Q fragments (16 x 64 -> two A frags over d halves), resident all loop
  v16bf qa[2];
  {
    const __bf16* p = Qp + (size_t)(qbase + lm) * HDM;
    #pragma unroll
    for (int dh = 0; dh < 2; ++dh) {
      const __bf16* pp = p + dh * 32 + hl * 8;
      qa[dh] = cat8(*(const v8bf*)pp, *(const v8bf*)(pp + 16));
    }
  }

  // all-ones B fragment: rowsum(P) = P x 1 via one extra WMMA per tile
  v16bf onesf;
  #pragma unroll
  for (int j = 0; j < 16; ++j) onesf[j] = f2bf(1.0f);

  v8f o[4] = {};
  float mrow[8], lrow[8];
  #pragma unroll
  for (int i = 0; i < 8; ++i) { mrow[i] = -1e30f; lrow[i] = 0.f; }

  const float scale = 0.5f;            // reference quirk: 1/sqrt(B=4)
  const float log2e = 1.44269504f;
  const int kvr = t >> 3, kvd = (t & 7) * 8;  // cooperative 32x64 staging

  for (int kb = 0; kb < SSEQ; kb += 32) {
    if (kb + 32 < SSEQ) {
      __builtin_prefetch(Kp + (size_t)(kb + 32 + kvr) * HDM + kvd, 0, 1);
      __builtin_prefetch(Vp + (size_t)(kb + 32 + kvr) * HDM + kvd, 0, 1);
    }
    __syncthreads();
#if HAVE_TDM
    if (w == 0) {
      // TDM: DMA the 32x64 bf16 K tile (row stride HDM elems) into sK,
      // padding +4 DWORDs every 32 DWORDs -> LDS row stride 72 bf16.
      const unsigned lds_base = (unsigned)(size_t)(&sK[0]);
      const unsigned long long ga =
          (unsigned long long)(size_t)(Kp + (size_t)kb * HDM);
      u32x4 g0 = {};
      g0[0] = 1u;                                        // count = 1 descriptor
      g0[1] = lds_base;                                  // lds_addr
      g0[2] = (unsigned)(ga & 0xffffffffu);              // global_addr[31:0]
      g0[3] = (unsigned)((ga >> 32) & 0x01ffffffu)       // global_addr[56:32]
              | (2u << 30);                              // type = 2 (image)
      i32x8 g1 = {};
      g1[0] = (int)((1u << 16)        // data_size = 2 bytes
              | (1u << 20)            // pad_enable
              | (4u << 22)            // pad_interval: 2<<4 = 32 DWORDs
              | (3u << 25));          // pad_amount: 3+1 = 4 DWORDs
      g1[1] = (int)(64u << 16);       // tensor_dim0 = 64   (bits 79:48 lo16)
      g1[2] = (int)(1024u << 16);     // tensor_dim1 = 1024 (bits 111:80 lo16)
      g1[3] = (int)(64u << 16);       // tile_dim0 = 64     (bits 127:112)
      g1[4] = 32;                     // tile_dim1 = 32     (bits 143:128)
      g1[5] = HDM;                    // tensor_dim0_stride (bits 207:160 lo32)
      i32x4 zz = {};
#if defined(__clang_major__) && (__clang_major__ >= 23)
      i32x8 z8 = {};
      __builtin_amdgcn_tensor_load_to_lds(g0, g1, zz, zz, z8, 0);
#else
      __builtin_amdgcn_tensor_load_to_lds(g0, g1, zz, zz, 0);
#endif
      __builtin_amdgcn_s_wait_tensorcnt(0);
    }
#else
    {
      const v8bf kv = *(const v8bf*)(Kp + (size_t)(kb + kvr) * HDM + kvd);
      *(v8bf*)&sK[kvr * 72 + kvd] = kv;
    }
#endif
    {
      const v8bf vv = *(const v8bf*)(Vp + (size_t)(kb + kvr) * HDM + kvd);
      #pragma unroll
      for (int j = 0; j < 8; ++j) sVt[(kvd + j) * 40 + kvr] = vv[j];
    }
    __syncthreads();

    // ---- scores S = Q K^T : 16 x 32, contraction over d = 64 ----
    v8f s[2] = {};
    #pragma unroll
    for (int c = 0; c < 2; ++c) {
      const __bf16* kbse = &sK[(c * 16 + lm) * 72 + hl * 8];
      v16bf b0 = cat8(*(const v8bf*)kbse,        *(const v8bf*)(kbse + 16));
      v16bf b1 = cat8(*(const v8bf*)(kbse + 32), *(const v8bf*)(kbse + 48));
      s[c] = wmma_bf16(qa[0], b0, s[c]);
      s[c] = wmma_bf16(qa[1], b1, s[c]);
    }
    #pragma unroll
    for (int c = 0; c < 2; ++c)
      #pragma unroll
      for (int vg = 0; vg < 8; ++vg) s[c][vg] *= scale;

    // ---- online softmax max (row = vg + 8*hl, spread over a 16-lane half) ----
    float psc[8];
    #pragma unroll
    for (int vg = 0; vg < 8; ++vg) {
      float v = fmaxf(s[0][vg], s[1][vg]);
      #pragma unroll
      for (int msk = 1; msk < 16; msk <<= 1) v = fmaxf(v, __shfl_xor(v, msk, 32));
      const float mnew = fmaxf(mrow[vg], v);
      psc[vg] = exp2f((mrow[vg] - mnew) * log2e);
      mrow[vg] = mnew;
      s[0][vg] = exp2f((s[0][vg] - mnew) * log2e);
      s[1][vg] = exp2f((s[1][vg] - mnew) * log2e);
    }
    #pragma unroll
    for (int c2 = 0; c2 < 4; ++c2)
      #pragma unroll
      for (int vg = 0; vg < 8; ++vg) o[c2][vg] *= psc[vg];

    // ---- P: C-layout -> A-layout via per-wave LDS bounce ----
    __bf16* pw = &sP[w * 16 * 40];
    #pragma unroll
    for (int c = 0; c < 2; ++c)
      #pragma unroll
      for (int vg = 0; vg < 8; ++vg)
        pw[(vg + 8 * hl) * 40 + c * 16 + lm] = f2bf(s[c][vg]);
    const __bf16* pr = &pw[lm * 40 + hl * 8];
    v16bf pa = cat8(*(const v8bf*)pr, *(const v8bf*)(pr + 16));

    // ---- row-sum via WMMA against ones: every lane gets its rows' sums ----
    v8f rs = {};
    rs = wmma_bf16(pa, onesf, rs);
    #pragma unroll
    for (int vg = 0; vg < 8; ++vg) lrow[vg] = lrow[vg] * psc[vg] + rs[vg];

    // ---- O += P V : contraction over 32 kv rows ----
    #pragma unroll
    for (int c2 = 0; c2 < 4; ++c2) {
      const __bf16* vb = &sVt[(c2 * 16 + lm) * 40 + hl * 8];
      v16bf vf = cat8(*(const v8bf*)vb, *(const v8bf*)(vb + 16));
      o[c2] = wmma_bf16(pa, vf, o[c2]);
    }
  }

  // ---- normalize + store ctx (B*S, HD) at column h*64 ----
  #pragma unroll
  for (int vg = 0; vg < 8; ++vg) {
    const float inv = 1.0f / lrow[vg];
    const int row = qbase + vg + 8 * hl;
    #pragma unroll
    for (int c2 = 0; c2 < 4; ++c2)
      Cp[(size_t)row * HDM + c2 * 16 + lm] = f2bf(o[c2][vg] * inv);
  }
}

// ---------------------------------------------------------------------------
extern "C" void kernel_launch(void* const* d_in, const int* in_sizes, int n_in,
                              void* d_out, int out_size, void* d_ws, size_t ws_size,
                              hipStream_t stream) {
  const float* query = (const float*)d_in[0];
  const float* key   = (const float*)d_in[1];
  const float* value = (const float*)d_in[2];
  const float* W_q   = (const float*)d_in[3];
  const float* b_q   = (const float*)d_in[4];
  const float* W_k   = (const float*)d_in[5];
  const float* b_k   = (const float*)d_in[6];
  const float* W_v   = (const float*)d_in[7];
  const float* b_v   = (const float*)d_in[8];
  const float* W_o   = (const float*)d_in[9];
  const float* b_o   = (const float*)d_in[10];

  const int M = BSZ * SSEQ;                  // 4096
  const size_t tile = (size_t)M * HDM;       // 4M elements
  __bf16* Qbf = (__bf16*)d_ws;               // 8 MB each, 32 MB total
  __bf16* Kbf = Qbf + tile;
  __bf16* Vbf = Kbf + tile;
  __bf16* Ctx = Vbf + tile;

  dim3 blk(256);
  dim3 gg(HDM / 128, M / 128);

  gemm_bias_kernel<false, true><<<gg, blk, 0, stream>>>(
      (const void*)query, W_q, b_q, (void*)Qbf, M, DMOD, HDM);
  gemm_bias_kernel<false, true><<<gg, blk, 0, stream>>>(
      (const void*)key, W_k, b_k, (void*)Kbf, M, DMOD, HDM);
  gemm_bias_kernel<false, true><<<gg, blk, 0, stream>>>(
      (const void*)value, W_v, b_v, (void*)Vbf, M, DMOD, HDM);

  dim3 ag(SSEQ / 128, BSZ * H_NUM);
  flash_attn_kernel<<<ag, blk, 0, stream>>>(Qbf, Kbf, Vbf, Ctx);

  gemm_bias_kernel<true, false><<<dim3(DMOD / 128, M / 128), blk, 0, stream>>>(
      (const void*)Ctx, W_o, b_o, d_out, M, HDM, DMOD);
}
